// MSTAGAT_Net_41704132444433
// MI455X (gfx1250) — matrix operand order repaired and checked by reading, not compiled, not run on hardware
//
#include <hip/hip_runtime.h>
#include <math.h>

// ---------------- problem constants ----------------
#define BB     8
#define TT     32
#define NN     1500
#define NP     1504      // N padded to multiple of 16/32 for WMMA tiles
#define HID    32
#define BOT    8
#define CC     16
#define HEADS  4
#define HD     8
#define SCALES 4
#define HZN    12
#define BH     (BB * HID)   // 256: diffusion RHS width (batch folded into columns)

typedef __bf16 bf16_t;
typedef __attribute__((ext_vector_type(16))) __bf16 v16bf;
typedef __attribute__((ext_vector_type(8)))  float  v8f;
typedef __attribute__((ext_vector_type(4)))  int    v4i;

// CDNA5 async global->LDS path (ASYNCcnt-tracked), if this toolchain declares it.
#if defined(__gfx1250__) && __has_builtin(__builtin_amdgcn_global_load_async_to_lds_b128) && __has_builtin(__builtin_amdgcn_s_wait_asynccnt)
#define HAVE_ASYNC_LDS 1
#define AS1 __attribute__((address_space(1)))
#define AS3 __attribute__((address_space(3)))
#endif

// ---------------- wave32 helpers ----------------
__device__ __forceinline__ float wsum32(float v) {
#pragma unroll
  for (int o = 16; o > 0; o >>= 1) v += __shfl_xor(v, o, 32);
  return v;
}
// layer-norm across the 32 lanes of a wave (lane = channel)
__device__ __forceinline__ float lnorm32(float x, float g, float b) {
  float m = wsum32(x) * (1.0f / 32.0f);
  float d = x - m;
  float var = wsum32(d * d) * (1.0f / 32.0f);
  return d * rsqrtf(var + 1e-5f) * g + b;
}
__device__ __forceinline__ void load8(const float* p, float* r) {
  float4 a = *(const float4*)p;
  float4 b = *(const float4*)(p + 4);
  r[0] = a.x; r[1] = a.y; r[2] = a.z; r[3] = a.w;
  r[4] = b.x; r[5] = b.y; r[6] = b.z; r[7] = b.w;
}

// ---------------- K1: temporal feature extraction ----------------
__global__ __launch_bounds__(256) void k_temporal(
    const float* __restrict__ x, const float* __restrict__ dw_w,
    const float* __restrict__ dw_b, const float* __restrict__ bn1_g,
    const float* __restrict__ bn1_b, const float* __restrict__ pw_w,
    const float* __restrict__ pw_b, const float* __restrict__ bn2_g,
    const float* __restrict__ bn2_b, const float* __restrict__ feat_w,
    const float* __restrict__ feat_b,
    float* __restrict__ h0, bf16_t* __restrict__ Hb) {
  int idx = blockIdx.x * blockDim.x + threadIdx.x;
  if (idx >= BB * NN) return;
  int b = idx / NN, n = idx % NN;
  float s[TT + 2];
  s[0] = 0.f; s[TT + 1] = 0.f;
  for (int t = 0; t < TT; ++t) s[t + 1] = x[((size_t)b * TT + t) * NN + n];
  float w0 = dw_w[0], w1 = dw_w[1], w2 = dw_w[2];
  float db = dw_b[0], g1 = bn1_g[0], b1 = bn1_b[0];
  float dwa[TT];
#pragma unroll
  for (int t = 0; t < TT; ++t) {
    float d = s[t] * w0 + s[t + 1] * w1 + s[t + 2] * w2;
    d = (d + db) * g1 + b1;
    dwa[t] = fmaxf(d, 0.f);
  }
  float mc[CC];
#pragma unroll
  for (int c = 0; c < CC; ++c) {
    float pwc = pw_w[c], pb = pw_b[c], g2 = bn2_g[c], b2 = bn2_b[c];
    float acc = 0.f;
    for (int t = 0; t < TT; ++t) {
      float p = (dwa[t] * pwc + pb) * g2 + b2;
      acc += fmaxf(p, 0.f);
    }
    mc[c] = acc * (1.0f / TT);
  }
  size_t row = (size_t)b * NN + n;
  for (int j = 0; j < HID; ++j) {
    float hv = feat_b[j];
#pragma unroll
    for (int c = 0; c < CC; ++c) hv += mc[c] * feat_w[c * HID + j];
    h0[row * HID + j] = hv;
    Hb[(size_t)n * BH + b * HID + j] = (bf16_t)hv;
  }
}

// ---------------- K2: row-normalized adjacency (bf16, padded) ----------------
__global__ __launch_bounds__(256) void k_build_ahat(
    const float* __restrict__ adj, bf16_t* __restrict__ Ahat) {
  __shared__ float red[256];
  int n = blockIdx.x;
  float s = 0.f;
  for (int m = threadIdx.x; m < NN; m += 256) s += adj[(size_t)n * NN + m];
  red[threadIdx.x] = s;
  __syncthreads();
  for (int o = 128; o > 0; o >>= 1) {
    if (threadIdx.x < o) red[threadIdx.x] += red[threadIdx.x + o];
    __syncthreads();
  }
  float inv = 1.0f / (red[0] + 1.0f + 1e-8f);  // +1 from identity diagonal
  for (int m = threadIdx.x; m < NN; m += 256) {
    float v = adj[(size_t)n * NN + m] + (m == n ? 1.0f : 0.0f);
    Ahat[(size_t)n * NP + m] = (bf16_t)(v * inv);
  }
}

// ---------------- K3: bf16 WMMA GEMM  C[M,Nc] = A[M,K] @ B[K,Nc] ----------------
// 256 threads = 8 waves (4 in M x 2 in N); each wave owns 16x32 -> 2 WMMAs/K-step
// with the A fragment reused. A tile staged via async global->LDS (ASYNCcnt);
// B tile transposed into LDS so every fragment is two contiguous ds_load_b128.
// Requires: Nc multiple of 64, K multiple of 32; M arbitrary (store-guarded).
#define TM 64
#define TN 64
#define TK 32
__global__ __launch_bounds__(256) void k_gemm_bf16(
    const bf16_t* __restrict__ A, int lda,
    const bf16_t* __restrict__ Bm, int ldb,
    float* __restrict__ Cm, bf16_t* __restrict__ Cbf, int ldc,
    int M, int Ncols, int Kd) {
  __shared__ __align__(16) bf16_t sA[TM][TK];   // 4 KB
  __shared__ __align__(16) bf16_t sBt[TN][TK];  // 4 KB, transposed [n][k]
  int tid = threadIdx.x;
  int wave = tid >> 5, lane = tid & 31;
  int wm = wave & 3, wn = wave >> 2;
  int l = lane & 15, hi = lane >> 4;
  int m0 = blockIdx.x * TM;
  int n0 = blockIdx.y * TN;

  v8f acc0 = {}, acc1 = {};
  int arow = wm * 16 + l;

  // A loader coords (straight-line: clamp OOB rows; their outputs are never stored)
  int la_r = tid >> 2;
  int la_c = (tid & 3) * 8;
  int la_gr = m0 + la_r; if (la_gr >= M) la_gr = M - 1;
  const bf16_t* aptr = A + (size_t)la_gr * lda + la_c;
  // B loader coords: 32 rows x 64 cols, 8 halves per thread, transposed store
  int lb_r = tid >> 3;
  int lb_c = (tid & 7) * 8;
  const bf16_t* bptr = Bm + (size_t)lb_r * ldb + n0 + lb_c;

  for (int k0 = 0; k0 < Kd; k0 += TK) {
#if HAVE_ASYNC_LDS
    {
      // builtin signature (probe-confirmed by diagnostics): param 0 is a
      // global (AS1) pointer to a 4 x i32 vector; LDS side mirrors it in AS3.
      v4i* gsrc_gen = (v4i*)(void*)const_cast<bf16_t*>(aptr + k0);
      v4i* ldst_gen = (v4i*)(void*)&sA[la_r][la_c];
      __builtin_amdgcn_global_load_async_to_lds_b128(
          (AS1 v4i*)gsrc_gen, (AS3 v4i*)ldst_gen, 0, 0);
    }
#else
    *(uint4*)(&sA[la_r][la_c]) = *(const uint4*)(aptr + k0);
#endif
    __builtin_prefetch(aptr + k0 + TK, 0, 1);  // speculative, arch-safe if OOB
    {
      union { uint4 u; bf16_t h[8]; } tb;
      tb.u = *(const uint4*)(bptr + (size_t)k0 * ldb);
#pragma unroll
      for (int i = 0; i < 8; ++i) sBt[lb_c + i][lb_r] = tb.h[i];
      __builtin_prefetch(bptr + (size_t)(k0 + TK) * ldb, 0, 1);
    }
#if HAVE_ASYNC_LDS
    __builtin_amdgcn_s_wait_asynccnt(0);
#endif
    __syncthreads();
    // A fragment (16x32 bf16): half h -> K = (h%8) + (h/8)*16 + hi*8
    // -> two contiguous 8-half chunks at K = hi*8 and K = 16 + hi*8.
    union { uint4 u[2]; v16bf v; } ua, ub0, ub1;
    ua.u[0] = *(const uint4*)(&sA[arow][hi * 8]);
    ua.u[1] = *(const uint4*)(&sA[arow][hi * 8 + 16]);
    // B fragment (32x16 bf16): lane -> col, half h -> K = (h%8) + (h/8)*16 + hi*8 (same as A)
    ub0.u[0] = *(const uint4*)(&sBt[wn * 32 + l][hi * 8]);
    ub0.u[1] = *(const uint4*)(&sBt[wn * 32 + l][hi * 8 + 16]);
    ub1.u[0] = *(const uint4*)(&sBt[wn * 32 + 16 + l][hi * 8]);
    ub1.u[1] = *(const uint4*)(&sBt[wn * 32 + 16 + l][hi * 8 + 16]);
    acc0 = __builtin_amdgcn_wmma_f32_16x16x32_bf16(
        false, ua.v, false, ub0.v, (short)0, acc0, false, false);
    acc1 = __builtin_amdgcn_wmma_f32_16x16x32_bf16(
        false, ua.v, false, ub1.v, (short)0, acc1, false, false);
    __syncthreads();
  }
#pragma unroll
  for (int nt = 0; nt < 2; ++nt) {
    int gcol = n0 + wn * 32 + nt * 16 + l;
#pragma unroll
    for (int r = 0; r < 8; ++r) {  // C layout: VGPR r -> M = r + hi*8
      int grow = m0 + wm * 16 + hi * 8 + r;
      if (grow < M && gcol < Ncols) {
        float v = (nt == 0) ? acc0[r] : acc1[r];
        Cm[(size_t)grow * ldc + gcol] = v;
        if (Cbf) Cbf[(size_t)grow * ldc + gcol] = (bf16_t)v;
      }
    }
  }
}

// ---------------- K4: per-scale transform + LN + fused sum + bottleneck + LN ----------------
__global__ __launch_bounds__(256) void k_fuse(
    const float* __restrict__ h0,
    const float* __restrict__ agg1, const float* __restrict__ agg2,
    const float* __restrict__ agg3,
    const float* __restrict__ scale_w, const float* __restrict__ scale_b,
    const float* __restrict__ ln_g, const float* __restrict__ ln_b,
    const float* __restrict__ fusw,
    const float* __restrict__ flo_w, const float* __restrict__ flo_b,
    const float* __restrict__ fhi_w, const float* __restrict__ fhi_b,
    const float* __restrict__ sp_g, const float* __restrict__ sp_b,
    float* __restrict__ h1) {
  int gw = blockIdx.x * (blockDim.x >> 5) + (threadIdx.x >> 5);
  int j = threadIdx.x & 31;
  if (gw >= BB * NN) return;
  int b = gw / NN, n = gw % NN;
  float f0 = fusw[0], f1 = fusw[1], f2 = fusw[2], f3 = fusw[3];
  float fm = fmaxf(fmaxf(f0, f1), fmaxf(f2, f3));
  float e0 = __expf(f0 - fm), e1 = __expf(f1 - fm);
  float e2 = __expf(f2 - fm), e3 = __expf(f3 - fm);
  float inv = 1.0f / (e0 + e1 + e2 + e3);
  float alpha[SCALES] = {e0 * inv, e1 * inv, e2 * inv, e3 * inv};

  size_t row = (size_t)gw;
  float h0v = h0[row * HID + j];
  float fused = 0.f;
#pragma unroll
  for (int i = 0; i < SCALES; ++i) {
    float a;
    if (i == 0) a = h0v;
    else {
      const float* ag = (i == 1) ? agg1 : (i == 2) ? agg2 : agg3;
      a = ag[(size_t)n * BH + b * HID + j];
    }
    float t = scale_b[i * HID + j];
    for (int k = 0; k < HID; ++k)
      t += __shfl(a, k, 32) * scale_w[((size_t)i * HID + k) * HID + j];
    t = lnorm32(t, ln_g[i * HID + j], ln_b[i * HID + j]);
    fused += alpha[i] * fmaxf(t, 0.f);
  }
  int p = j & (BOT - 1);
  float lo = flo_b[p];
  for (int k = 0; k < HID; ++k) lo += __shfl(fused, k, 32) * flo_w[k * BOT + p];
  float o = fhi_b[j];
#pragma unroll
  for (int q = 0; q < BOT; ++q) o += __shfl(lo, q, 32) * fhi_w[q * HID + j];
  float r = o + h0v;
  h1[row * HID + j] = lnorm32(r, sp_g[j], sp_b[j]);
}

// ---------------- K5: qkv bottleneck (32 -> 24 -> 96), wave per row ----------------
__global__ __launch_bounds__(256) void k_qkv(
    const float* __restrict__ h1,
    const float* __restrict__ lo_w, const float* __restrict__ lo_b,
    const float* __restrict__ hi_w, const float* __restrict__ hi_b,
    float* __restrict__ qkv) {
  int gw = blockIdx.x * (blockDim.x >> 5) + (threadIdx.x >> 5);
  int j = threadIdx.x & 31;
  if (gw >= BB * NN) return;
  size_t row = (size_t)gw;
  float hv = h1[row * HID + j];
  int p = (j < 24) ? j : (j - 24);
  float lov = lo_b[p];
  for (int k = 0; k < HID; ++k) lov += __shfl(hv, k, 32) * lo_w[k * 24 + p];
#pragma unroll
  for (int rpt = 0; rpt < 3; ++rpt) {
    int col = j + rpt * 32;
    float o = hi_b[col];
#pragma unroll
    for (int k = 0; k < 24; ++k) o += __shfl(lov, k, 32) * hi_w[k * 96 + col];
    qkv[row * 96 + col] = o;
  }
}

// ---------------- K6: struct = U@V + sigmoid(mix)*scale*adj ----------------
__global__ __launch_bounds__(256) void k_struct(
    const float* __restrict__ U, const float* __restrict__ V,
    const float* __restrict__ adj, const float* __restrict__ adj_mix,
    const float* __restrict__ adj_scale, float* __restrict__ sm) {
  size_t idx = (size_t)blockIdx.x * blockDim.x + threadIdx.x;
  if (idx >= (size_t)NN * NN) return;
  int n = (int)(idx / NN), m = (int)(idx % NN);
  float s = 0.f;
#pragma unroll
  for (int d = 0; d < BOT; ++d) s += U[n * BOT + d] * V[(size_t)d * NN + m];
  float sig = 1.0f / (1.0f + __expf(-adj_mix[0]));
  sm[idx] = s + sig * adj_scale[0] * adj[idx];
}

// ---------------- K7: flash attention, wave per (b, head, query row) ----------------
__global__ __launch_bounds__(256) void k_attn(
    const float* __restrict__ qkv, const float* __restrict__ sm,
    float* __restrict__ oat) {
  int gw = blockIdx.x * (blockDim.x >> 5) + (threadIdx.x >> 5);
  int lane = threadIdx.x & 31;
  if (gw >= BB * HEADS * NN) return;
  int b = gw / (HEADS * NN);
  int rem = gw % (HEADS * NN);
  int hh = rem / NN;
  int n = rem % NN;
  const float qs = 0.35355339059327373f;  // 1/sqrt(8)
  float q[HD];
  load8(qkv + ((size_t)b * NN + n) * 96 + hh * HD, q);
#pragma unroll
  for (int d = 0; d < HD; ++d) q[d] *= qs;
  float mmax = -3.0e38f, ssum = 0.f;
  float acc[HD];
#pragma unroll
  for (int d = 0; d < HD; ++d) acc[d] = 0.f;
  const float* srow = sm + (size_t)n * NN;
  for (int m = lane; m < NN; m += 32) {
    const float* base = qkv + ((size_t)b * NN + m) * 96 + hh * HD;
    float kk[HD], vv[HD];
    load8(base + HID, kk);       // keys at +32
    load8(base + 2 * HID, vv);   // values at +64
    float s = srow[m];
#pragma unroll
    for (int d = 0; d < HD; ++d) s += q[d] * kk[d];
    float nm = fmaxf(mmax, s);
    float sc = __expf(mmax - nm);
    float p = __expf(s - nm);
    ssum = ssum * sc + p;
#pragma unroll
    for (int d = 0; d < HD; ++d) acc[d] = acc[d] * sc + p * vv[d];
    mmax = nm;
  }
#pragma unroll
  for (int o = 16; o > 0; o >>= 1) {  // merge 32 lane-local softmaxes
    float om = __shfl_xor(mmax, o, 32);
    float os = __shfl_xor(ssum, o, 32);
    float nm = fmaxf(mmax, om);
    float s1 = __expf(mmax - nm), s2 = __expf(om - nm);
    ssum = ssum * s1 + os * s2;
#pragma unroll
    for (int d = 0; d < HD; ++d) {
      float oa = __shfl_xor(acc[d], o, 32);
      acc[d] = acc[d] * s1 + oa * s2;
    }
    mmax = nm;
  }
  if (lane == 0) {
    float invs = 1.0f / ssum;
    float* dst = oat + ((size_t)b * NN + n) * HID + hh * HD;
#pragma unroll
    for (int d = 0; d < HD; ++d) dst[d] = acc[d] * invs;
  }
}

// ---------------- K8: attention out-proj (32->8->32) + residual + LN ----------------
__global__ __launch_bounds__(256) void k_outproj(
    const float* __restrict__ o_in, const float* __restrict__ h1,
    const float* __restrict__ lo_w, const float* __restrict__ lo_b,
    const float* __restrict__ hi_w, const float* __restrict__ hi_b,
    const float* __restrict__ g, const float* __restrict__ bp,
    float* __restrict__ h2) {
  int gw = blockIdx.x * (blockDim.x >> 5) + (threadIdx.x >> 5);
  int j = threadIdx.x & 31;
  if (gw >= BB * NN) return;
  size_t row = (size_t)gw;
  float ov = o_in[row * HID + j];
  int p = j & (BOT - 1);
  float lo = lo_b[p];
  for (int k = 0; k < HID; ++k) lo += __shfl(ov, k, 32) * lo_w[k * BOT + p];
  float oo = hi_b[j];
#pragma unroll
  for (int q = 0; q < BOT; ++q) oo += __shfl(lo, q, 32) * hi_w[q * HID + j];
  float r = oo + h1[row * HID + j];
  h2[row * HID + j] = lnorm32(r, g[j], bp[j]);
}

// ---------------- K9: gated horizon predictor ----------------
__global__ __launch_bounds__(256) void k_pred(
    const float* __restrict__ h2, const float* __restrict__ x,
    const float* __restrict__ lo_w, const float* __restrict__ lo_b,
    const float* __restrict__ mid_g, const float* __restrict__ mid_b,
    const float* __restrict__ hi_w, const float* __restrict__ hi_b,
    const float* __restrict__ log_dec,
    const float* __restrict__ gw1, const float* __restrict__ gb1,
    const float* __restrict__ gw2, const float* __restrict__ gb2,
    float* __restrict__ out) {
  int idx = blockIdx.x * blockDim.x + threadIdx.x;
  if (idx >= BB * NN) return;
  int b = idx / NN, n = idx % NN;
  float h[HID];
  for (int k = 0; k < HID; ++k) h[k] = h2[(size_t)idx * HID + k];
  float lo[BOT], g1[BOT];
#pragma unroll
  for (int p = 0; p < BOT; ++p) {
    float a = lo_b[p], c = gb1[p];
    for (int k = 0; k < HID; ++k) {
      a += h[k] * lo_w[k * BOT + p];
      c += h[k] * gw1[k * BOT + p];
    }
    lo[p] = a;
    g1[p] = fmaxf(c, 0.f);
  }
  float m = 0.f;
#pragma unroll
  for (int p = 0; p < BOT; ++p) m += lo[p];
  m *= (1.0f / BOT);
  float var = 0.f;
#pragma unroll
  for (int p = 0; p < BOT; ++p) { float d = lo[p] - m; var += d * d; }
  var *= (1.0f / BOT);
  float rs = rsqrtf(var + 1e-5f);
  float xm[BOT];
#pragma unroll
  for (int p = 0; p < BOT; ++p)
    xm[p] = fmaxf((lo[p] - m) * rs * mid_g[p] + mid_b[p], 0.f);
  float xlast = x[((size_t)b * TT + (TT - 1)) * NN + n];
  float dec = __expf(log_dec[0]);
#pragma unroll
  for (int t = 0; t < HZN; ++t) {
    float ip = hi_b[t], gl = gb2[t];
#pragma unroll
    for (int p = 0; p < BOT; ++p) {
      ip += xm[p] * hi_w[p * HZN + t];
      gl += g1[p] * gw2[p * HZN + t];
    }
    float gate = 1.0f / (1.0f + __expf(-gl));
    float prog = xlast * __expf(-dec * (float)(t + 1));
    out[(size_t)idx * HZN + t] = gate * ip + (1.0f - gate) * prog;
  }
}

// ---------------- launcher ----------------
extern "C" void kernel_launch(void* const* d_in, const int* in_sizes, int n_in,
                              void* d_out, int out_size, void* d_ws, size_t ws_size,
                              hipStream_t stream) {
  (void)in_sizes; (void)n_in; (void)out_size; (void)ws_size;
  const float* x       = (const float*)d_in[0];
  const float* adj     = (const float*)d_in[1];
  const float* dw_w    = (const float*)d_in[2];
  const float* dw_b    = (const float*)d_in[3];
  const float* bn1_g   = (const float*)d_in[4];
  const float* bn1_b   = (const float*)d_in[5];
  const float* pw_w    = (const float*)d_in[6];
  const float* pw_b    = (const float*)d_in[7];
  const float* bn2_g   = (const float*)d_in[8];
  const float* bn2_b   = (const float*)d_in[9];
  const float* feat_w  = (const float*)d_in[10];
  const float* feat_b  = (const float*)d_in[11];
  const float* scale_w = (const float*)d_in[12];
  const float* scale_b = (const float*)d_in[13];
  const float* sln_g   = (const float*)d_in[14];
  const float* sln_b   = (const float*)d_in[15];
  const float* fusw    = (const float*)d_in[16];
  const float* flo_w   = (const float*)d_in[17];
  const float* flo_b   = (const float*)d_in[18];
  const float* fhi_w   = (const float*)d_in[19];
  const float* fhi_b   = (const float*)d_in[20];
  const float* sp_g    = (const float*)d_in[21];
  const float* sp_b    = (const float*)d_in[22];
  const float* qlo_w   = (const float*)d_in[23];
  const float* qlo_b   = (const float*)d_in[24];
  const float* qhi_w   = (const float*)d_in[25];
  const float* qhi_b   = (const float*)d_in[26];
  const float* gU      = (const float*)d_in[27];
  const float* gV      = (const float*)d_in[28];
  const float* adj_mix = (const float*)d_in[29];
  const float* adj_sc  = (const float*)d_in[30];
  const float* olo_w   = (const float*)d_in[31];
  const float* olo_b   = (const float*)d_in[32];
  const float* ohi_w   = (const float*)d_in[33];
  const float* ohi_b   = (const float*)d_in[34];
  const float* at_g    = (const float*)d_in[35];
  const float* at_b    = (const float*)d_in[36];
  const float* plo_w   = (const float*)d_in[37];
  const float* plo_b   = (const float*)d_in[38];
  const float* mid_g   = (const float*)d_in[39];
  const float* mid_b   = (const float*)d_in[40];
  const float* phi_w   = (const float*)d_in[41];
  const float* phi_b   = (const float*)d_in[42];
  const float* log_dec = (const float*)d_in[43];
  const float* gw1     = (const float*)d_in[44];
  const float* gb1     = (const float*)d_in[45];
  const float* gw2     = (const float*)d_in[46];
  const float* gb2     = (const float*)d_in[47];
  float* out = (float*)d_out;

  // workspace bump allocator (256B aligned)
  char* ws = (char*)d_ws;
  size_t off = 0;
  auto alloc = [&](size_t bytes) -> void* {
    void* p = ws + off;
    off = (off + bytes + 255) & ~(size_t)255;
    return p;
  };
  float*  h0    = (float*) alloc((size_t)BB * NN * HID * 4);
  bf16_t* Hb    = (bf16_t*)alloc((size_t)NP * BH * 2);
  bf16_t* Ahat  = (bf16_t*)alloc((size_t)NP * NP * 2);
  float*  agg1  = (float*) alloc((size_t)NP * BH * 4);
  float*  agg2  = (float*) alloc((size_t)NP * BH * 4);
  float*  agg3  = (float*) alloc((size_t)NP * BH * 4);
  bf16_t* agg1b = (bf16_t*)alloc((size_t)NP * BH * 2);
  bf16_t* agg2b = (bf16_t*)alloc((size_t)NP * BH * 2);
  float*  h1    = (float*) alloc((size_t)BB * NN * HID * 4);
  float*  qkv   = (float*) alloc((size_t)BB * NN * 96 * 4);
  float*  smat  = (float*) alloc((size_t)NN * NN * 4);
  float*  oat   = (float*) alloc((size_t)BB * NN * HID * 4);
  float*  h2    = (float*) alloc((size_t)BB * NN * HID * 4);

  // zero bf16 pads so padded GEMM rows/cols contribute nothing
  (void)hipMemsetAsync(Hb,   0, (size_t)NP * BH * 2, stream);
  (void)hipMemsetAsync(Ahat, 0, (size_t)NP * NP * 2, stream);

  k_temporal<<<(BB * NN + 255) / 256, 256, 0, stream>>>(
      x, dw_w, dw_b, bn1_g, bn1_b, pw_w, pw_b, bn2_g, bn2_b, feat_w, feat_b,
      h0, Hb);
  k_build_ahat<<<NN, 256, 0, stream>>>(adj, Ahat);

  // agg_i = A_hat^i @ H  (iterated; never forms A_hat^2 / A_hat^3 explicitly)
  dim3 gg((NP + TM - 1) / TM, BH / TN);
  k_gemm_bf16<<<gg, 256, 0, stream>>>(Ahat, NP, Hb,    BH, agg1, agg1b, BH, NP, BH, NP);
  k_gemm_bf16<<<gg, 256, 0, stream>>>(Ahat, NP, agg1b, BH, agg2, agg2b, BH, NP, BH, NP);
  k_gemm_bf16<<<gg, 256, 0, stream>>>(Ahat, NP, agg2b, BH, agg3, (bf16_t*)nullptr, BH, NP, BH, NP);

  k_fuse<<<(BB * NN) / 8, 256, 0, stream>>>(
      h0, agg1, agg2, agg3, scale_w, scale_b, sln_g, sln_b, fusw,
      flo_w, flo_b, fhi_w, fhi_b, sp_g, sp_b, h1);
  k_qkv<<<(BB * NN) / 8, 256, 0, stream>>>(h1, qlo_w, qlo_b, qhi_w, qhi_b, qkv);
  k_struct<<<(int)(((size_t)NN * NN + 255) / 256), 256, 0, stream>>>(
      gU, gV, adj, adj_mix, adj_sc, smat);
  k_attn<<<(BB * HEADS * NN) / 8, 256, 0, stream>>>(qkv, smat, oat);
  k_outproj<<<(BB * NN) / 8, 256, 0, stream>>>(
      oat, h1, olo_w, olo_b, ohi_w, ohi_b, at_g, at_b, h2);
  k_pred<<<(BB * NN + 255) / 256, 256, 0, stream>>>(
      h2, x, plo_w, plo_b, mid_g, mid_b, phi_w, phi_b, log_dec,
      gw1, gb1, gw2, gb2, out);
}
